// MultiHeadedAttention_76261439307828
// MI455X (gfx1250) — matrix operand order repaired and checked by reading, not compiled, not run on hardware
//
#include <hip/hip_runtime.h>
#include <hip/hip_bf16.h>

// ---------------- problem constants ----------------
constexpr int B_  = 16;
constexpr int S_  = 256;
constexpr int D_  = 1024;
constexpr int HP_ = 16;        // positional heads, dk = 64
constexpr int HC_ = 8;         // channel heads,   dk = 32
constexpr int DK_ = D_ / HP_;  // 64
constexpr int SK_ = S_ / HC_;  // 32
constexpr int M1_ = B_ * S_;   // 4096 rows for positional path
constexpr int M2_ = B_ * D_;   // 16384 rows for channel path
constexpr float EPS_ = 1e-6f;

// ---------------- WMMA types (gfx1250, wave32) ----------------
typedef __attribute__((ext_vector_type(16))) __bf16 v16bf;
typedef __attribute__((ext_vector_type(8)))  float  v8f;

union Frag16 { v16bf v; uint4 q[2]; };
union Pack8bf { __bf16 h[8]; uint4 q; };

__device__ inline v8f zero8() {
    v8f z;
#pragma unroll
    for (int i = 0; i < 8; ++i) z[i] = 0.0f;
    return z;
}

__device__ inline v8f wmma_bf16(v16bf a, v16bf b, v8f c) {
    return __builtin_amdgcn_wmma_f32_16x16x32_bf16(
        false, a, false, b, (short)0, c, false, false);
}
// A-frag (16x32 bf16) per-lane layout: e=0..7  -> K = 8*half + e
//                                      e=8..15 -> K = 16 + 8*half + (e-8)
// B-frag (32x16 bf16) per-lane layout: lane = N, element e -> same K as A-frag element e

// ---- CDNA5 async global->LDS DMA (ASYNCcnt-tracked, no VGPR staging) ----
__device__ inline void async_ld_b128(unsigned lds_byte_off, const void* gaddr) {
    asm volatile("global_load_async_to_lds_b128 %0, %1, off"
                 :: "v"(lds_byte_off), "v"(gaddr) : "memory");
}
__device__ inline void wait_async0() {
    asm volatile("s_wait_asynccnt 0x0" ::: "memory");
}

// ---------------- f32 -> bf16 transpose (per batch matrix) ----------------
__global__ __launch_bounds__(256) void cvt_transpose_kernel(
    const float* __restrict__ in, __bf16* __restrict__ out, int R, int C) {
    __shared__ float t[32][33];
    const int tx = threadIdx.x & 31, ty = threadIdx.x >> 5;   // ty in 0..7
    const int c0 = blockIdx.x * 32, r0 = blockIdx.y * 32;
    const size_t bo = (size_t)blockIdx.z * R * C;
#pragma unroll
    for (int i = 0; i < 4; ++i)
        t[ty + 8 * i][tx] = in[bo + (size_t)(r0 + ty + 8 * i) * C + c0 + tx];
    __syncthreads();
#pragma unroll
    for (int i = 0; i < 4; ++i)
        out[bo + (size_t)(c0 + ty + 8 * i) * R + r0 + tx] = (__bf16)t[tx][ty + 8 * i];
}

// ---------------- LayerNorm over channels (D), unbiased std ----------------
__global__ __launch_bounds__(256) void ln_p_kernel(const float* __restrict__ x,
                                                   const float* __restrict__ ap,
                                                   const float* __restrict__ bp,
                                                   __bf16* __restrict__ out) {
    __shared__ float s1[256], s2[256];
    const int row = blockIdx.x;   // b*S + s
    const int tid = threadIdx.x;
    const float* xr = x + (size_t)row * D_;
    float v[4], a = 0.f, sq = 0.f;
#pragma unroll
    for (int i = 0; i < 4; ++i) { v[i] = xr[tid + i * 256]; a += v[i]; sq += v[i] * v[i]; }
    s1[tid] = a; s2[tid] = sq;
    __syncthreads();
    for (int off = 128; off > 0; off >>= 1) {
        if (tid < off) { s1[tid] += s1[tid + off]; s2[tid] += s2[tid + off]; }
        __syncthreads();
    }
    float mean = s1[0] / (float)D_;
    float var  = fmaxf((s2[0] - (float)D_ * mean * mean) / (float)(D_ - 1), 0.f);
    float inv  = 1.0f / (sqrtf(var) + EPS_);
#pragma unroll
    for (int i = 0; i < 4; ++i) {
        int c = tid + i * 256;
        out[(size_t)row * D_ + c] = (__bf16)(ap[c] * (v[i] - mean) * inv + bp[c]);
    }
}

// ------- LayerNorm over sequence (S), coalesced, writes transposed [B,D,S] -------
// grid: (D/32, B); block 256 = 32 (d) x 8 (s-groups)
__global__ __launch_bounds__(256) void ln_c_kernel(const float* __restrict__ xatt,
                                                   const float* __restrict__ ac,
                                                   const float* __restrict__ bc0,
                                                   __bf16* __restrict__ outT) {
    __shared__ float ps1[8][32], ps2[8][32];
    __shared__ float mcol[32], icol[32];
    __shared__ __align__(16) __bf16 tile[32][264];   // 264*2B row stride (16B multiple)
    const int tid = threadIdx.x, tx = tid & 31, ty = tid >> 5;
    const int b = blockIdx.y, d0 = blockIdx.x * 32;

    float vals[32], s1 = 0.f, s2 = 0.f;
#pragma unroll
    for (int i = 0; i < 32; ++i) {
        const int s = ty * 32 + i;
        float v = xatt[((size_t)b * S_ + s) * D_ + d0 + tx];
        vals[i] = v; s1 += v; s2 += v * v;
    }
    ps1[ty][tx] = s1; ps2[ty][tx] = s2;
    __syncthreads();
    if (ty == 0) {
        float a = 0.f, q = 0.f;
#pragma unroll
        for (int k = 0; k < 8; ++k) { a += ps1[k][tx]; q += ps2[k][tx]; }
        float mean = a / (float)S_;
        float var  = fmaxf((q - (float)S_ * mean * mean) / (float)(S_ - 1), 0.f);
        mcol[tx] = mean;
        icol[tx] = 1.0f / (sqrtf(var) + EPS_);
    }
    __syncthreads();
    const float mean = mcol[tx], inv = icol[tx];
#pragma unroll
    for (int i = 0; i < 32; ++i) {
        const int s = ty * 32 + i;
        tile[tx][s] = (__bf16)(ac[s] * (vals[i] - mean) * inv + bc0[s]);
    }
    __syncthreads();
    const int r = tid >> 3, ch = (tid & 7) * 32;   // 32 bf16 = 64B per thread
    const uint4* src = (const uint4*)&tile[r][ch];
    uint4* dst = (uint4*)&outT[((size_t)b * D_ + d0 + r) * S_ + ch];
#pragma unroll
    for (int i = 0; i < 4; ++i) dst[i] = src[i];
}

// ---------------- Tiled WMMA GEMM:  C = A[MxK] * Wt[NxK]^T + bias ----------------
// Wt is the TRANSPOSED weight (N-major): A and B staging are symmetric.
// Double-buffered async global->LDS DMA overlaps the next tile with WMMAs.
// block: 256 threads (8 waves); tile 128(M) x 128(N); K step 32; 8 WMMAs/wave/step.
// OUTMODE 0: bf16 row-major [M,N]
// OUTMODE 1: bf16 transposed per batch-b: out[(b*N + n)*TOK + tok], m = b*TOK + tok
// OUTMODE 2: f32 final epilogue: out[(b*S_ + n)*D_ + d], m = b*D_ + d
template <int OUTMODE>
__global__ __launch_bounds__(256) void gemm_bf16_kernel(
    const __bf16* __restrict__ A, const __bf16* __restrict__ Wt,
    const float* __restrict__ bias, void* __restrict__ Out,
    int N, int K,
    size_t wBatchStride, size_t outBatchStride, int biasBatchStride, int tokShift)
{
    __shared__ __align__(16) __bf16 Ash[2][128][40];
    __shared__ __align__(16) __bf16 Bsh[2][128][40];

    const int tid  = threadIdx.x;
    const int lane = tid & 31, w = tid >> 5;
    const int ln16 = lane & 15, half = lane >> 4;
    const int mw = w & 3, nw = w >> 2;
    const int m0 = blockIdx.y * 128, n0 = blockIdx.x * 128;
    const int batch = blockIdx.z;

    const __bf16* Wb = Wt + wBatchStride * (size_t)batch;
    const float* biasb = bias + (size_t)biasBatchStride * batch;

    v8f acc[2][4];
#pragma unroll
    for (int mi = 0; mi < 2; ++mi)
#pragma unroll
        for (int j = 0; j < 4; ++j) acc[mi][j] = zero8();

    const int sr = tid >> 1, scb = (tid & 1) * 16;   // stage: 128 rows x 32, 32B/thread

    auto issue_tile = [&](int kb, int buf) {
        const unsigned ldsA = (unsigned)(uintptr_t)&Ash[buf][sr][scb];
        const __bf16* ga = A + (size_t)(m0 + sr) * K + kb * 32 + scb;
        async_ld_b128(ldsA,      ga);
        async_ld_b128(ldsA + 16, ga + 8);
        const unsigned ldsB = (unsigned)(uintptr_t)&Bsh[buf][sr][scb];
        const __bf16* gb = Wb + (size_t)(n0 + sr) * K + kb * 32 + scb;
        async_ld_b128(ldsB,      gb);
        async_ld_b128(ldsB + 16, gb + 8);
    };

    const int nkb = K / 32;
    issue_tile(0, 0);
    wait_async0();
    __syncthreads();

    int buf = 0;
    for (int kb = 0; kb < nkb; ++kb) {
        if (kb + 1 < nkb) issue_tile(kb + 1, buf ^ 1);   // DMA next tile during compute

        Frag16 af[2];
#pragma unroll
        for (int mi = 0; mi < 2; ++mi) {
            const int r = mw * 32 + mi * 16 + ln16;
            af[mi].q[0] = *(const uint4*)&Ash[buf][r][8 * half];
            af[mi].q[1] = *(const uint4*)&Ash[buf][r][16 + 8 * half];
        }
#pragma unroll
        for (int j = 0; j < 4; ++j) {
            const int n = nw * 64 + j * 16 + ln16;
            Frag16 bf_;
            bf_.q[0] = *(const uint4*)&Bsh[buf][n][8 * half];
            bf_.q[1] = *(const uint4*)&Bsh[buf][n][16 + 8 * half];
            acc[0][j] = wmma_bf16(af[0].v, bf_.v, acc[0][j]);
            acc[1][j] = wmma_bf16(af[1].v, bf_.v, acc[1][j]);
        }
        wait_async0();
        __syncthreads();
        buf ^= 1;
    }

#pragma unroll
    for (int mi = 0; mi < 2; ++mi) {
#pragma unroll
        for (int j = 0; j < 4; ++j) {
            const int n = n0 + nw * 64 + j * 16 + ln16;
            const float bv = biasb[n];
            const int mb = m0 + mw * 32 + mi * 16 + 8 * half;   // v=0 row; rows mb..mb+7
            if (OUTMODE == 0) {
#pragma unroll
                for (int v = 0; v < 8; ++v)
                    ((__bf16*)Out)[outBatchStride * batch + (size_t)(mb + v) * N + n] =
                        (__bf16)(acc[mi][j][v] + bv);
            } else if (OUTMODE == 1) {
                const int b = mb >> tokShift, tok0 = mb & ((1 << tokShift) - 1);
                Pack8bf p;
#pragma unroll
                for (int v = 0; v < 8; ++v) p.h[v] = (__bf16)(acc[mi][j][v] + bv);
                *(uint4*)&((__bf16*)Out)[(((size_t)b * N + n) << tokShift) + tok0] = p.q;
            } else {
                const int b = mb >> 10, d0 = mb & (D_ - 1);
                float* o = (float*)Out + ((size_t)b * S_ + n) * D_ + d0;
                float4 o0, o1;
                o0.x = acc[mi][j][0] + bv; o0.y = acc[mi][j][1] + bv;
                o0.z = acc[mi][j][2] + bv; o0.w = acc[mi][j][3] + bv;
                o1.x = acc[mi][j][4] + bv; o1.y = acc[mi][j][5] + bv;
                o1.z = acc[mi][j][6] + bv; o1.w = acc[mi][j][7] + bv;
                *(float4*)o       = o0;
                *(float4*)(o + 4) = o1;
            }
        }
    }
}

// ---------------- fused attention: one block = (b, head, 16-row q tile) ----------------
// Q,K row-major [b*NK + tok, C]; V TRANSPOSED: [(b*C + c)*NK + tok].
// 128 threads (4 waves). Scores in LDS (f32), block softmax, 1/rowsum in PV epilogue.
template <int NK, int DKH, int H, bool OUTF32>
__global__ __launch_bounds__(128) void attn_kernel(
    const __bf16* __restrict__ Q, const __bf16* __restrict__ Km,
    const __bf16* __restrict__ Vt, float* __restrict__ outF, __bf16* __restrict__ outB)
{
    constexpr int C   = H * DKH;
    constexpr int NKT = NK / 16;
    __shared__ float sc[16][NK];
    __shared__ float pm[16][8];
    __shared__ float rowmax[16];
    __shared__ float rowinv[16];

    const int tid  = threadIdx.x;
    const int lane = tid & 31, w = tid >> 5;
    const int ln16 = lane & 15, half = lane >> 4;
    const int qt = blockIdx.x, h = blockIdx.y, b = blockIdx.z;
    const size_t rowBase = (size_t)b * NK;
    const int colOff = h * DKH;
    const float scale = rsqrtf((float)DKH);

    // ---- scores = (Q K^T) * scale ----  (Q fragments hoisted: invariant over kt)
    Frag16 afq[DKH / 32];
#pragma unroll
    for (int kc = 0; kc < DKH / 32; ++kc) {
        const __bf16* qrow = Q + (rowBase + qt * 16 + ln16) * C + colOff + kc * 32;
        afq[kc].q[0] = *(const uint4*)(qrow + 8 * half);
        afq[kc].q[1] = *(const uint4*)(qrow + 16 + 8 * half);
    }
    for (int kt = w; kt < NKT; kt += 4) {
        v8f acc = zero8();
#pragma unroll
        for (int kc = 0; kc < DKH / 32; ++kc) {
            Frag16 bf_;
            const __bf16* krow = Km + (rowBase + kt * 16 + ln16) * C + colOff + kc * 32;
            bf_.q[0] = *(const uint4*)(krow + 8 * half);
            bf_.q[1] = *(const uint4*)(krow + 16 + 8 * half);
            acc = wmma_bf16(afq[kc].v, bf_.v, acc);
        }
#pragma unroll
        for (int v = 0; v < 8; ++v)
            sc[v + 8 * half][kt * 16 + ln16] = acc[v] * scale;
    }
    __syncthreads();

    // ---- softmax (max-subtract); keep unnormalized exp in LDS ----
    constexpr int CH = NK / 8;
    {
        const int r = tid & 15, c = tid >> 4;
        float m = -3.4e38f;
        for (int j = c * CH; j < (c + 1) * CH; ++j) m = fmaxf(m, sc[r][j]);
        pm[r][c] = m;
    }
    __syncthreads();
    if (tid < 16) {
        float m = pm[tid][0];
        for (int c = 1; c < 8; ++c) m = fmaxf(m, pm[tid][c]);
        rowmax[tid] = m;
    }
    __syncthreads();
    {
        const int r = tid & 15, c = tid >> 4;
        const float m = rowmax[r];
        float s = 0.f;
        for (int j = c * CH; j < (c + 1) * CH; ++j) {
            float e = __expf(sc[r][j] - m);
            sc[r][j] = e;
            s += e;
        }
        pm[r][c] = s;
    }
    __syncthreads();
    if (tid < 16) {
        float s = 0.f;
        for (int c = 0; c < 8; ++c) s += pm[tid][c];
        rowinv[tid] = 1.0f / s;
    }
    __syncthreads();

    // ---- PV: out = (exp_scores @ V) * rowinv ----
    constexpr int NT = DKH / 16;
    if (w < NT) {
        const int nt = w;
        v8f acc = zero8();
        for (int kc = 0; kc < NK / 32; ++kc) {
            const float* srow = &sc[ln16][kc * 32];
            float4 f0 = *(const float4*)(srow + 8 * half);
            float4 f1 = *(const float4*)(srow + 8 * half + 4);
            float4 f2 = *(const float4*)(srow + 16 + 8 * half);
            float4 f3 = *(const float4*)(srow + 16 + 8 * half + 4);
            Frag16 af;
            af.v[0] = (__bf16)f0.x; af.v[1] = (__bf16)f0.y; af.v[2] = (__bf16)f0.z; af.v[3] = (__bf16)f0.w;
            af.v[4] = (__bf16)f1.x; af.v[5] = (__bf16)f1.y; af.v[6] = (__bf16)f1.z; af.v[7] = (__bf16)f1.w;
            af.v[8]  = (__bf16)f2.x; af.v[9]  = (__bf16)f2.y; af.v[10] = (__bf16)f2.z; af.v[11] = (__bf16)f2.w;
            af.v[12] = (__bf16)f3.x; af.v[13] = (__bf16)f3.y; af.v[14] = (__bf16)f3.z; af.v[15] = (__bf16)f3.w;
            const __bf16* vrow = Vt + ((size_t)b * C + colOff + nt * 16 + ln16) * NK + kc * 32;
            Frag16 bf_;
            bf_.q[0] = *(const uint4*)(vrow + 8 * half);
            bf_.q[1] = *(const uint4*)(vrow + 16 + 8 * half);
            acc = wmma_bf16(af.v, bf_.v, acc);
        }
#pragma unroll
        for (int v = 0; v < 8; ++v) {
            const int r = v + 8 * half;
            const float val = acc[v] * rowinv[r];
            const size_t idx = (rowBase + qt * 16 + r) * (size_t)C + colOff + nt * 16 + ln16;
            if (OUTF32) outF[idx] = val;
            else        outB[idx] = (__bf16)val;
        }
    }
}

// ---------------- launcher ----------------
extern "C" void kernel_launch(void* const* d_in, const int* in_sizes, int n_in,
                              void* d_out, int out_size, void* d_ws, size_t ws_size,
                              hipStream_t stream) {
    (void)in_sizes; (void)n_in; (void)out_size; (void)ws_size;

    const float* x   = (const float*)d_in[0];
    const float* Wp  = (const float*)d_in[1];
    const float* bp  = (const float*)d_in[2];
    const float* Wc  = (const float*)d_in[3];
    const float* bc  = (const float*)d_in[4];
    const float* a_p = (const float*)d_in[5];
    const float* b_p = (const float*)d_in[6];
    const float* a_c = (const float*)d_in[7];
    const float* b_c = (const float*)d_in[8];
    float* out = (float*)d_out;

    char* ws = (char*)d_ws;
    size_t off = 0;
    auto take = [&](size_t bytes) { char* p = ws + off; off = (off + bytes + 255) & ~(size_t)255; return p; };

    __bf16* WpT_bf   = (__bf16*)take((size_t)3 * D_ * D_ * 2);   // transposed weights
    __bf16* WcT_bf   = (__bf16*)take((size_t)4 * S_ * S_ * 2);
    __bf16* xp_bf    = (__bf16*)take((size_t)M1_ * D_ * 2);
    __bf16* QKp_bf   = (__bf16*)take((size_t)2 * M1_ * D_ * 2);
    __bf16* Vpt_bf   = (__bf16*)take((size_t)M1_ * D_ * 2);      // transposed V
    float*  xatt     = (float*) take((size_t)M1_ * D_ * 4);
    __bf16* xcT_bf   = (__bf16*)take((size_t)M2_ * S_ * 2);
    __bf16* QKc_bf   = (__bf16*)take((size_t)2 * M2_ * S_ * 2);
    __bf16* Vct_bf   = (__bf16*)take((size_t)M2_ * S_ * 2);      // transposed V
    __bf16* xcatt_bf = (__bf16*)take((size_t)M2_ * S_ * 2);

    // 1) weights -> transposed bf16 (Wt[n][k])
    cvt_transpose_kernel<<<dim3(D_ / 32, D_ / 32, 3), dim3(256), 0, stream>>>(Wp, WpT_bf, D_, D_);
    cvt_transpose_kernel<<<dim3(S_ / 32, S_ / 32, 4), dim3(256), 0, stream>>>(Wc, WcT_bf, S_, S_);

    // 2) LayerNorm over channels -> xp (bf16)
    ln_p_kernel<<<dim3(M1_), dim3(256), 0, stream>>>(x, a_p, b_p, xp_bf);

    // 3) positional projections: Q,K row-major; V transposed (tokShift=8)
    gemm_bf16_kernel<0><<<dim3(D_ / 128, M1_ / 128, 2), dim3(256), 0, stream>>>(
        xp_bf, WpT_bf, bp, (void*)QKp_bf, D_, D_,
        (size_t)D_ * D_, (size_t)M1_ * D_, D_, 0);
    gemm_bf16_kernel<1><<<dim3(D_ / 128, M1_ / 128, 1), dim3(256), 0, stream>>>(
        xp_bf, WpT_bf + (size_t)2 * D_ * D_, bp + 2 * D_, (void*)Vpt_bf, D_, D_,
        0, 0, 0, 8);

    // 4) positional attention (NK=256, dk=64, 16 heads), f32 out
    attn_kernel<S_, DK_, HP_, true><<<dim3(S_ / 16, HP_, B_), dim3(128), 0, stream>>>(
        QKp_bf, QKp_bf + (size_t)M1_ * D_, Vpt_bf, xatt, nullptr);

    // 5) LayerNorm over sequence (coalesced) -> transposed [B,D,S] bf16
    ln_c_kernel<<<dim3(D_ / 32, B_), dim3(256), 0, stream>>>(xatt, a_c, b_c, xcT_bf);

    // 6) channel projections: Q,K row-major; V transposed (tokShift=10)
    gemm_bf16_kernel<0><<<dim3(S_ / 128, M2_ / 128, 2), dim3(256), 0, stream>>>(
        xcT_bf, WcT_bf, bc, (void*)QKc_bf, S_, S_,
        (size_t)S_ * S_, (size_t)M2_ * S_, S_, 0);
    gemm_bf16_kernel<1><<<dim3(S_ / 128, M2_ / 128, 1), dim3(256), 0, stream>>>(
        xcT_bf, WcT_bf + (size_t)2 * S_ * S_, bc + 2 * S_, (void*)Vct_bf, S_, S_,
        0, 0, 0, 10);

    // 7) channel attention (NK=1024 tokens, dk=32, 8 heads), bf16 out
    attn_kernel<D_, SK_, HC_, false><<<dim3(D_ / 16, HC_, B_), dim3(128), 0, stream>>>(
        QKc_bf, QKc_bf + (size_t)M2_ * S_, Vct_bf, nullptr, xcatt_bf);

    // 8) output projection with transposed f32 epilogue -> d_out [B,S,D]
    gemm_bf16_kernel<2><<<dim3(S_ / 128, M2_ / 128, 1), dim3(256), 0, stream>>>(
        xcatt_bf, WcT_bf + (size_t)3 * S_ * S_, bc + 3 * S_, (void*)out, S_, S_,
        0, 0, 0, 0);
}
